// VNN_cell_3015067041878
// MI455X (gfx1250) — matrix operand order repaired and checked by reading, not compiled, not run on hardware
//
#include <hip/hip_runtime.h>
#include <hip/hip_bf16.h>
#include <stdint.h>

typedef __attribute__((ext_vector_type(16))) __bf16 v16bf;
typedef __attribute__((ext_vector_type(8)))  float  v8f;
typedef __attribute__((ext_vector_type(4)))  unsigned int u32x4;
typedef __attribute__((ext_vector_type(8)))  int          i32x8;
typedef __attribute__((ext_vector_type(4)))  int          i32x4;

#define NW    16          // waves per workgroup (wave32)
#define TPB   (NW * 32)   // 512 threads
#define TILES 4           // 16-row M-tiles per wave -> 16*4*16 = 1024 rows
#define BATCH 1024

#if __has_builtin(__builtin_amdgcn_tensor_load_to_lds) && \
    __has_builtin(__builtin_amdgcn_s_wait_tensorcnt)
#define HAS_TDM 1
#else
#define HAS_TDM 0
#endif

__device__ __forceinline__ float mishf(float x) {
    float sp = (x > 20.0f) ? x : log1pf(expf(x));
    return x * tanhf(sp);
}

#if HAS_TDM
// TDM: DMA W[g] ([Kact rows x 10 cols] f32, contiguous) from global into LDS.
// D# built per CDNA5 ISA 8.3/8.4 (2D tensor, data_size=4B, dim0=tile0=10,
// dim1=tile1=Kact, dim0_stride=10). Issued by one wave; TENSORcnt tracks it.
__device__ __forceinline__ void tdm_load_w(const float* gW, float* ldsDst, int Kact) {
    uint64_t ga = (uint64_t)(uintptr_t)gW;
    uint32_t la = (uint32_t)(uintptr_t)ldsDst;   // generic LDS ptr: low 32 bits = LDS offset
    u32x4 g0;
    g0[0] = 1u;                                            // count=1, user mode, no gather
    g0[1] = la;                                            // lds_addr
    g0[2] = (uint32_t)(ga & 0xFFFFFFFFu);                  // global_addr[31:0]
    g0[3] = (uint32_t)((ga >> 32) & 0x01FFFFFFu) | (2u << 30); // addr[56:32] | type=2
    i32x8 g1;
    g1[0] = (2 << 16);                 // data_size = 4 bytes; no multicast/pad/iterate
    g1[1] = (10 << 16);                // tensor_dim0 = 10
    g1[2] = (Kact & 0xFFFF) << 16;     // tensor_dim1 = Kact
    g1[3] = (10 << 16);                // tile_dim0 = 10
    g1[4] = Kact;                      // tile_dim1 = Kact
    g1[5] = 10;                        // tensor_dim0_stride = 10
    g1[6] = 0;
    g1[7] = 0;
    i32x4 z4 = {0, 0, 0, 0};
    i32x8 z8 = {0, 0, 0, 0, 0, 0, 0, 0};
    // 6-arg form (clang-23 / therock-10.0 lane)
    __builtin_amdgcn_tensor_load_to_lds(g0, g1, z4, z4, z8, 0);
}
#endif

// One workgroup == one DAG group: BN(mish(X[:,g,:] @ W[g] + b[g])) over all
// 1024 batch rows, so BatchNorm stats reduce entirely in LDS.
// Requires padded K (KSTEPS*32); Kact is the true K (<= KSTEPS*32).
template <int KSTEPS>
__global__ __launch_bounds__(TPB)
void vnn_level_kernel(const float* __restrict__ x,      // [B, xBstride] padded input
                      const float* __restrict__ W,      // [L, Kact, 10]
                      const float* __restrict__ bias,   // [L, 10]
                      const float* __restrict__ gamma,  // [L, 10]
                      const float* __restrict__ beta,   // [L, 10]
                      float*       __restrict__ out,
                      int Kact, int xBstride, int xGroupStride,
                      int outBstride, int outChild, int outGroupPad)
{
    constexpr int K = KSTEPS * 32;
    __shared__ __bf16 Wl[16 * K];          // bf16 weights, [n][k]: B-frags contiguous
#if HAS_TDM
    __shared__ float  Wraw[K * 10];        // raw f32 W tile landed by the TDM
#endif
    __shared__ float  psum[TPB], psq[TPB];
    __shared__ float  scaleS[16], shiftS[16];

    const int g    = blockIdx.x;
    const int tid  = threadIdx.x;
    const int lane = tid & 31;
    const int wave = tid >> 5;
    const int col  = lane & 15;            // N index (and M sub-row for A frags)
    const int hi   = lane >> 4;            // half-wave select
    const size_t wBase = (size_t)g * Kact * 10;

    // ---- Stage W[g] into LDS as bf16, padded N->16 and K->Kpad with zeros ----
#if HAS_TDM
    if (wave == 0) {
        tdm_load_w(W + wBase, Wraw, Kact);
        __builtin_amdgcn_s_wait_tensorcnt(0);
    }
    __syncthreads();
    for (int idx = tid; idx < 16 * K; idx += TPB) {
        int n = idx / K, k = idx % K;
        Wl[idx] = (__bf16)((n < 10 && k < Kact) ? Wraw[k * 10 + n] : 0.0f);
    }
#else
    for (int idx = tid; idx < 16 * K; idx += TPB) {
        int n = idx / K, k = idx % K;
        Wl[idx] = (__bf16)((n < 10 && k < Kact) ? W[wBase + (size_t)k * 10 + n] : 0.0f);
    }
#endif
    __syncthreads();

    const float bv = (col < 10) ? bias[g * 10 + col] : 0.0f;

    v8f acc[TILES];
    const v8f vzero = {0.f, 0.f, 0.f, 0.f, 0.f, 0.f, 0.f, 0.f};
#pragma unroll
    for (int t = 0; t < TILES; ++t) acc[t] = vzero;

    const int koffA = hi ? 8 : 0;          // A-frag lane K offset (ISA 16x32 layout)

#pragma unroll
    for (int ks = 0; ks < KSTEPS; ++ks) {
        const int kbase = ks << 5;

        // B fragment: 32x16 bf16; lanes 0-15 hold K=kbase+0..7,16..23, lanes 16-31 +8..15,24..31
        v16bf bf;
        {
            const int kb = col * K + kbase + (hi ? 8 : 0);
#pragma unroll
            for (int e = 0; e < 16; ++e) bf[e] = Wl[kb + e + (e < 8 ? 0 : 8)];
        }

#pragma unroll
        for (int t = 0; t < TILES; ++t) {
            const int arow = ((wave * TILES + t) << 4) + col;   // batch row
            const float* xp = x + (size_t)arow * xBstride
                                + (size_t)g * xGroupStride + kbase + koffA;
            const float4 f0 = ((const float4*)xp)[0];
            const float4 f1 = ((const float4*)xp)[1];
            const float4 f2 = ((const float4*)(xp + 16))[0];
            const float4 f3 = ((const float4*)(xp + 16))[1];
            if (ks + 1 < KSTEPS)           // global_prefetch_b8, keep-near
                __builtin_prefetch(xp + 32, 0, 3);
            v16bf af;
            af[0]=(__bf16)f0.x; af[1]=(__bf16)f0.y; af[2]=(__bf16)f0.z; af[3]=(__bf16)f0.w;
            af[4]=(__bf16)f1.x; af[5]=(__bf16)f1.y; af[6]=(__bf16)f1.z; af[7]=(__bf16)f1.w;
            af[8]=(__bf16)f2.x; af[9]=(__bf16)f2.y; af[10]=(__bf16)f2.z; af[11]=(__bf16)f2.w;
            af[12]=(__bf16)f3.x; af[13]=(__bf16)f3.y; af[14]=(__bf16)f3.z; af[15]=(__bf16)f3.w;
            acc[t] = __builtin_amdgcn_wmma_f32_16x16x32_bf16(
                false, af, false, bf, (short)0, acc[t], false, false);
        }
    }

    // ---- bias + mish, per-lane BN partials (every lane owns one N column) ----
    float lsum = 0.0f, lsq = 0.0f;
#pragma unroll
    for (int t = 0; t < TILES; ++t)
#pragma unroll
        for (int i = 0; i < 8; ++i) {
            float v = mishf(acc[t][i] + bv);
            acc[t][i] = v;
            lsum += v; lsq += v * v;
        }
    psum[tid] = lsum; psq[tid] = lsq;
    __syncthreads();

    // ---- finish BN stats: 16 threads, one per output column ----
    if (tid < 16) {
        float s = 0.0f, q = 0.0f;
        for (int w = 0; w < NW; ++w) {
            s += psum[w * 32 + tid] + psum[w * 32 + tid + 16];
            q += psq [w * 32 + tid] + psq [w * 32 + tid + 16];
        }
        const float inv = 1.0f / (float)BATCH;
        float mean = s * inv;
        float var  = q * inv - mean * mean;
        float rstd = rsqrtf(var + 1e-5f);
        float gm = (tid < 10) ? gamma[g * 10 + tid] : 1.0f;
        float bt = (tid < 10) ? beta [g * 10 + tid] : 0.0f;
        scaleS[tid] = rstd * gm;
        shiftS[tid] = bt - mean * rstd * gm;
    }
    __syncthreads();

    // ---- affine + store (C/D layout: VGPR i -> M = i + 8*hi, N = col) ----
    const float sc = scaleS[col], sh = shiftS[col];
    const int gBase = (g / outChild) * outGroupPad + (g % outChild) * 10;
    if (col < 10) {
#pragma unroll
        for (int t = 0; t < TILES; ++t) {
            const int m0 = ((wave * TILES + t) << 4) + (hi ? 8 : 0);
#pragma unroll
            for (int i = 0; i < 8; ++i)
                out[(size_t)(m0 + i) * outBstride + gBase + col] = acc[t][i] * sc + sh;
        }
    }
}

extern "C" void kernel_launch(void* const* d_in, const int* in_sizes, int n_in,
                              void* d_out, int out_size, void* d_ws, size_t ws_size,
                              hipStream_t stream) {
    (void)in_sizes; (void)n_in; (void)out_size; (void)ws_size;
    const float* features = (const float*)d_in[0];
    const float* W1  = (const float*)d_in[1];
    const float* b1  = (const float*)d_in[2];
    const float* g1  = (const float*)d_in[3];
    const float* bt1 = (const float*)d_in[4];
    const float* W2  = (const float*)d_in[5];
    const float* b2  = (const float*)d_in[6];
    const float* g2  = (const float*)d_in[7];
    const float* bt2 = (const float*)d_in[8];
    const float* W3  = (const float*)d_in[9];
    const float* b3  = (const float*)d_in[10];
    const float* g3  = (const float*)d_in[11];
    const float* bt3 = (const float*)d_in[12];

    // Zero-padded intermediates so every GEMM sees K % 32 == 0 (no tail code).
    const size_t h1pBytes = (size_t)BATCH * 20 * 128 * 4;   // [B, 20, 128]
    const size_t h2pBytes = (size_t)BATCH * 224 * 4;        // [B, 224]
    float* h1p = (float*)d_ws;
    float* h2p = (float*)((char*)d_ws + h1pBytes);
    (void)hipMemsetAsync(d_ws, 0, h1pBytes + h2pBytes, stream);

    // Level 1: 200 leaf groups, K=160 (=5*32), features [B, 32000].
    //   out -> h1p[b][g/10][ (g%10)*10 + col ]  (mid-group-padded layout)
    vnn_level_kernel<5><<<200, TPB, 0, stream>>>(
        features, W1, b1, g1, bt1, h1p,
        /*Kact*/160, /*xB*/32000, /*xGrpStride*/160,
        /*outB*/2560, /*outChild*/10, /*outGroupPad*/128);

    // Level 2: 20 mid groups, Kact=100 padded to 128; x = h1p.
    vnn_level_kernel<4><<<20, TPB, 0, stream>>>(
        h1p, W2, b2, g2, bt2, h2p,
        /*Kact*/100, /*xB*/2560, /*xGrpStride*/128,
        /*outB*/224, /*outChild*/20, /*outGroupPad*/224);

    // Root: 1 group, Kact=200 padded to 224; x = h2p; out = d_out [B,10].
    vnn_level_kernel<7><<<1, TPB, 0, stream>>>(
        h2p, W3, b3, g3, bt3, (float*)d_out,
        /*Kact*/200, /*xB*/224, /*xGrpStride*/224,
        /*outB*/10, /*outChild*/1, /*outGroupPad*/10);
}
